// PartialNLL_35476429864987
// MI455X (gfx1250) — compile-verified, matrix operand order checked by reading, not run on hardware
//
#include <hip/hip_runtime.h>
#include <math.h>

typedef __attribute__((ext_vector_type(16))) _Float16 v16h;
typedef __attribute__((ext_vector_type(8)))  _Float16 v8h;
typedef __attribute__((ext_vector_type(8)))  float    v8f;

// ---------------------------------------------------------------------------
// Kernel 1: split exp(theta[j]) into f16 hi/lo tables (hi + lo ~= f32 exact).
// Row-independent, so this hoists ALL f16 conversion work out of the N^2 loop.
// ---------------------------------------------------------------------------
__global__ __launch_bounds__(256)
void cox_prep_exp_35476429864987(const float* __restrict__ theta,
                                 _Float16* __restrict__ ehi,
                                 _Float16* __restrict__ elo, int n) {
  int i = blockIdx.x * 256 + threadIdx.x;
  if (i < n) {
    float e = expf(theta[i]);
    _Float16 hi = (_Float16)e;
    ehi[i] = hi;
    elo[i] = (_Float16)(e - (float)hi);
  }
}

// ---------------------------------------------------------------------------
// Kernel 2: per-16-row tile, risk[i] = sum_j [t_j >= t_i] * exp(theta_j)
// via V_WMMA_F32_16X16X32_F16 row-sums: A = masked exp (hi/lo f16 pair),
// B = all-ones  ->  D[i][c] = sum_K A[i][K], layout-permutation robust.
// Two independent accumulators so the two WMMAs per chunk can overlap.
// One wave (32 lanes) per 16-row tile; 8 waves per block.
// ---------------------------------------------------------------------------
__global__ __launch_bounds__(256)
void cox_risk_wmma_35476429864987(const float* __restrict__ theta,
                                  const float* __restrict__ tv,
                                  const int*   __restrict__ obs,
                                  const _Float16* __restrict__ ehi,
                                  const _Float16* __restrict__ elo,
                                  float* __restrict__ partial, int n) {
  const int lane = threadIdx.x & 31;
  const int wave = blockIdx.x * 8 + (threadIdx.x >> 5);
  const int i0   = wave * 16;
  const int g    = lane >> 4;             // lane group: K-partition selector
  const float ti = tv[i0 + (lane & 15)];  // this lane's A-row time

  v8f acc_hi = {};
  v8f acc_lo = {};
  v16h ones;
#pragma unroll
  for (int h = 0; h < 16; ++h) ones[h] = (_Float16)1.0f;
  const _Float16 zero_h = (_Float16)0.0f;

  for (int j0 = 0; j0 < n; j0 += 32) {
    // This lane-group's 16 K-slots cover two contiguous 8-runs of j.
    const int b1 = j0 + 8 * g;        // run 1: 8 elements
    const int b2 = j0 + 16 + 8 * g;   // run 2: 8 elements

    if (j0 + 32 < n) {
      __builtin_prefetch(tv + b1 + 32, 0, 0);
      __builtin_prefetch(ehi + b1 + 32, 0, 0);
      __builtin_prefetch(elo + b1 + 32, 0, 0);
    }

    // Times stay f32 (mask >= must match reference bit-exactly).
    float4 t0 = *(const float4*)(tv + b1);
    float4 t1 = *(const float4*)(tv + b1 + 4);
    float4 t2 = *(const float4*)(tv + b2);
    float4 t3 = *(const float4*)(tv + b2 + 4);
    // Pre-split f16 exp tables: one b128 per 8-run.
    v8h eh0 = *(const v8h*)(ehi + b1);
    v8h eh1 = *(const v8h*)(ehi + b2);
    v8h el0 = *(const v8h*)(elo + b1);
    v8h el1 = *(const v8h*)(elo + b2);

    float tvv[16] = {t0.x, t0.y, t0.z, t0.w, t1.x, t1.y, t1.z, t1.w,
                     t2.x, t2.y, t2.z, t2.w, t3.x, t3.y, t3.z, t3.w};

    v16h ahi, alo;
#pragma unroll
    for (int h = 0; h < 8; ++h) {
      bool m0 = tvv[h]     >= ti;
      bool m1 = tvv[h + 8] >= ti;
      ahi[h]     = m0 ? eh0[h] : zero_h;
      ahi[h + 8] = m1 ? eh1[h] : zero_h;
      alo[h]     = m0 ? el0[h] : zero_h;
      alo[h + 8] = m1 ? el1[h] : zero_h;
    }

    // D = A x ones + C  ->  per-row K-sum, f32 accumulate.
    acc_hi = __builtin_amdgcn_wmma_f32_16x16x32_f16(
        false, ahi, false, ones, (short)0, acc_hi, false, false);
    acc_lo = __builtin_amdgcn_wmma_f32_16x16x32_f16(
        false, alo, false, ones, (short)0, acc_lo, false, false);
  }

  // Column 0 of D lives in lane 0 (rows i0..i0+7 in acc[0..7]) and
  // lane 16 (rows i0+8..i0+15). Fixed-order scalar sums -> deterministic.
  if ((lane & 15) == 0) {
    float s = 0.0f;
#pragma unroll
    for (int p = 0; p < 8; ++p) {
      int r = i0 + 8 * g + p;
      float risk = acc_hi[p] + acc_lo[p];
      if (obs[r] != 0) s += theta[r] - logf(risk);
    }
    partial[wave * 2 + g] = s;
  }
}

// ---------------------------------------------------------------------------
// Kernel 3: deterministic final reduction (single block, fixed tree order).
// ---------------------------------------------------------------------------
__global__ __launch_bounds__(256)
void cox_finalize_35476429864987(const float* __restrict__ partial, int nPartial,
                                 const int* __restrict__ obs, int n,
                                 float* __restrict__ out) {
  __shared__ float sf[256];
  __shared__ int   si[256];
  int t = threadIdx.x;
  float s = 0.0f;
  for (int k = t; k < nPartial; k += 256) s += partial[k];
  int c = 0;
  for (int k = t; k < n; k += 256) c += obs[k];
  sf[t] = s;
  si[t] = c;
  __syncthreads();
  for (int off = 128; off > 0; off >>= 1) {
    if (t < off) { sf[t] += sf[t + off]; si[t] += si[t + off]; }
    __syncthreads();
  }
  if (t == 0) out[0] = -sf[0] / (float)si[0];
}

// ---------------------------------------------------------------------------
extern "C" void kernel_launch(void* const* d_in, const int* in_sizes, int n_in,
                              void* d_out, int out_size, void* d_ws, size_t ws_size,
                              hipStream_t stream) {
  const float* theta = (const float*)d_in[0];   // (N,1) f32
  const float* timev = (const float*)d_in[1];   // (N,)  f32
  const int*   obs   = (const int*)d_in[2];     // (N,)  i32
  float* out = (float*)d_out;

  const int n = in_sizes[0];                    // 16384

  // Workspace layout (16-byte aligned slices):
  _Float16* ehi     = (_Float16*)d_ws;          // n halves
  _Float16* elo     = ehi + n;                  // n halves
  float*    partial = (float*)(elo + n);        // 2*(n/16) floats

  const int nTiles   = n / 16;                  // 16-row tiles (one wave each)
  const int nPartial = nTiles * 2;

  cox_prep_exp_35476429864987<<<(n + 255) / 256, 256, 0, stream>>>(
      theta, ehi, elo, n);
  cox_risk_wmma_35476429864987<<<nTiles / 8, 256, 0, stream>>>(
      theta, timev, obs, ehi, elo, partial, n);
  cox_finalize_35476429864987<<<1, 256, 0, stream>>>(partial, nPartial, obs, n, out);
}
